// LSTMModel_39960375722571
// MI455X (gfx1250) — compile-verified, matrix operand order checked by reading, not run on hardware
//
#include <hip/hip_runtime.h>

// ---------------- CDNA5 (gfx1250) fused 2-layer LSTM + head ----------------
// wave32; WMMA 16x16x32 bf16 (A/B bf16, C/D f32 accumulate)
// - 16 batch rows per workgroup, 4 waves split the 4H=256 gate columns
// - weights live in LDS as bf16 with bank-conflict-free row padding
// - h staging double-buffered in LDS (padded rows)
// - cell state + biases live in VGPRs; WMMA chains start from inline-0 C

typedef __bf16 v16bf  __attribute__((ext_vector_type(16)));
typedef __bf16 bf16x8 __attribute__((ext_vector_type(8)));
typedef float  v8f    __attribute__((ext_vector_type(8)));

#define T_STEPS 512
#define D_IN    9
#define HDIM    64
#define GDIM    256   // 4*H

// Padded LDS row strides (in bf16 elements).
// WSTR = 72 halfs = 36 dwords: 36*n mod 64 distinct for n=0..15 -> conflict-free
// fragment loads across the 16 lanes of a half; 144 B rows stay 16-B aligned.
// XSTR = 56 halfs = 28 dwords: same property for the K<=32 Wih0 tile.
#define WSTR 72
#define XSTR 56

__device__ __forceinline__ float sigmoid_f(float x) {
    return 1.0f / (1.0f + __expf(-x));
}
__device__ __forceinline__ float tanh_f(float x) {
    // tanh(x) = 1 - 2/(exp(2x)+1)
    return 1.0f - 2.0f / (__expf(2.0f * x) + 1.0f);
}

// Load one fragment (16 bf16 per lane) from a row-major row.
// Per ISA 7.12.2 (16-bit A 16x32): lane half j<8 -> k = koff+j,
// j>=8 -> k = koff+16+(j-8). koff includes kblock*32 + lane-half base.
__device__ __forceinline__ v16bf ld_frag(const __bf16* __restrict__ row, int koff) {
    bf16x8 lo = *(const bf16x8*)(row + koff);
    bf16x8 hi = *(const bf16x8*)(row + koff + 16);
    v16bf r;
#pragma unroll
    for (int i = 0; i < 8; ++i) { r[i] = lo[i]; r[i + 8] = hi[i]; }
    return r;
}

__device__ __forceinline__ v8f wmma_bf16(v16bf a, v16bf b, v8f c) {
    return __builtin_amdgcn_wmma_f32_16x16x32_bf16(
        /*neg_a=*/false, a, /*neg_b=*/false, b,
        /*c_mod=*/(short)0, c, /*reuse_a=*/false, /*reuse_b=*/false);
}

__global__ void __launch_bounds__(128)
lstm2_fused_kernel(const float* __restrict__ x,
                   const float* __restrict__ Wih0, const float* __restrict__ Whh0,
                   const float* __restrict__ bih0, const float* __restrict__ bhh0,
                   const float* __restrict__ Wih1, const float* __restrict__ Whh1,
                   const float* __restrict__ bih1, const float* __restrict__ bhh1,
                   const float* __restrict__ fc1w, const float* __restrict__ fc1b,
                   const float* __restrict__ fc2w, const float* __restrict__ fc2b,
                   float* __restrict__ out)
{
    // bf16 weights staged once in LDS (~148 KB of the 320 KB WGP LDS)
    __shared__ __align__(16) __bf16 sWih0[GDIM * XSTR];
    __shared__ __align__(16) __bf16 sWhh0[GDIM * WSTR];
    __shared__ __align__(16) __bf16 sWih1[GDIM * WSTR];
    __shared__ __align__(16) __bf16 sWhh1[GDIM * WSTR];
    __shared__ __align__(16) __bf16 sH0[2][16 * WSTR];     // double-buffered h staging
    __shared__ __align__(16) __bf16 sH1[2][16 * WSTR];

    const int tid  = threadIdx.x;
    const int lane = tid & 31;
    const int wave = tid >> 5;            // 0..3: which 16-wide column group of H
    const int rowBase = blockIdx.x * 16;  // 16 batch rows per workgroup

    // ---- stage weights (fp32 -> bf16, padded rows) ----
    for (int i = tid; i < GDIM * 32; i += 128) {
        int g = i >> 5, k = i & 31;
        sWih0[g * XSTR + k] = (k < D_IN) ? (__bf16)Wih0[g * D_IN + k] : (__bf16)0.0f;
    }
    for (int i = tid; i < GDIM * HDIM; i += 128) {
        int g = i >> 6, k = i & 63;
        sWhh0[g * WSTR + k] = (__bf16)Whh0[i];
        sWih1[g * WSTR + k] = (__bf16)Wih1[i];
        sWhh1[g * WSTR + k] = (__bf16)Whh1[i];
    }
    for (int i = tid; i < 16 * WSTR; i += 128) {   // h0 = h1 = 0 at t=0
        sH0[0][i] = (__bf16)0.0f;
        sH1[0][i] = (__bf16)0.0f;
    }
    __syncthreads();

    const int m16   = lane & 15;               // A row / B row / D column
    const int hbase = (lane & 16) ? 8 : 0;     // per-lane-half K base
    const int mrow  = (lane & 16) ? 8 : 0;     // D-frag row offset

    // Per-lane gate biases (b_ih + b_hh), per gate type; held in VGPRs for all T.
    float b0g[4], b1g[4];
#pragma unroll
    for (int gt = 0; gt < 4; ++gt) {
        const int col = gt * HDIM + wave * 16 + m16;   // gate column in [0,256)
        b0g[gt] = bih0[col] + bhh0[col];
        b1g[gt] = bih1[col] + bhh1[col];
    }

    // cell states: one 16x16 f32 tile per layer (D-frag layout), in VGPRs
    v8f c0 = {}; v8f c1 = {};

    // x A-fragment template: zeros beyond k = D_IN-1 (K padded to 32)
    v16bf xzero;
#pragma unroll
    for (int j = 0; j < 16; ++j) xzero[j] = (__bf16)0.0f;

    const float* xrow = x + (size_t)(rowBase + m16) * T_STEPS * D_IN;

    auto load_xfrag = [&](int t) -> v16bf {
        v16bf a = xzero;
        const float* xp = xrow + (size_t)t * D_IN;
        if (lane < 16) {
#pragma unroll
            for (int j = 0; j < 8; ++j) a[j] = (__bf16)xp[j];   // k = 0..7
        } else {
            a[0] = (__bf16)xp[8];                               // k = 8
        }
        return a;
    };

    v16bf xf = load_xfrag(0);
    int p = 0;
    v8f acc[4];
    const v8f zero8 = {};   // WMMA C starts from inline 0

    for (int t = 0; t < T_STEPS; ++t) {
        // ================= layer 0 =================
        const __bf16* h0row = &sH0[p][m16 * WSTR];
        v16bf a0 = ld_frag(h0row, hbase);
        v16bf a1 = ld_frag(h0row, 32 + hbase);
#pragma unroll
        for (int gt = 0; gt < 4; ++gt) {           // i, f, g, o
            const int nt = gt * 4 + wave;          // N-tile of gates
            const __bf16* bih = &sWih0[(nt * 16 + m16) * XSTR];
            const __bf16* bhh = &sWhh0[(nt * 16 + m16) * WSTR];
            v8f c = wmma_bf16(xf, ld_frag(bih, hbase), zero8);  // x @ Wih0^T
            c = wmma_bf16(a0, ld_frag(bhh, hbase), c);          // h @ Whh0^T
            c = wmma_bf16(a1, ld_frag(bhh, 32 + hbase), c);
            acc[gt] = c;
        }
        // prefetch next timestep's x while gates settle
        v16bf xfn = load_xfrag((t + 1 < T_STEPS) ? (t + 1) : t);
        // warm the x stream well ahead of the serial recurrence
        __builtin_prefetch(xrow + (size_t)((t + 8 < T_STEPS) ? t + 8 : t) * D_IN, 0, 1);

        __bf16* h0out = &sH0[p ^ 1][0];
#pragma unroll
        for (int r = 0; r < 8; ++r) {
            float ig = sigmoid_f(acc[0][r] + b0g[0]);
            float fg = sigmoid_f(acc[1][r] + b0g[1]);
            float gg = tanh_f(acc[2][r] + b0g[2]);
            float og = sigmoid_f(acc[3][r] + b0g[3]);
            float cn = fg * c0[r] + ig * gg;
            c0[r] = cn;
            float hn = og * tanh_f(cn);
            h0out[(r + mrow) * WSTR + wave * 16 + m16] = (__bf16)hn;
        }
        __syncthreads();   // new h0 visible to all waves

        // ================= layer 1 =================
        const __bf16* x1row = &sH0[p ^ 1][m16 * WSTR];   // input = fresh h0
        v16bf b0x = ld_frag(x1row, hbase);
        v16bf b1x = ld_frag(x1row, 32 + hbase);
        const __bf16* h1row = &sH1[p][m16 * WSTR];
        v16bf a0h = ld_frag(h1row, hbase);
        v16bf a1h = ld_frag(h1row, 32 + hbase);
#pragma unroll
        for (int gt = 0; gt < 4; ++gt) {
            const int nt = gt * 4 + wave;
            const __bf16* bih = &sWih1[(nt * 16 + m16) * WSTR];
            const __bf16* bhh = &sWhh1[(nt * 16 + m16) * WSTR];
            v8f c = wmma_bf16(b0x, ld_frag(bih, hbase), zero8); // h0 @ Wih1^T
            c = wmma_bf16(b1x, ld_frag(bih, 32 + hbase), c);
            c = wmma_bf16(a0h, ld_frag(bhh, hbase), c);         // h1 @ Whh1^T
            c = wmma_bf16(a1h, ld_frag(bhh, 32 + hbase), c);
            acc[gt] = c;
        }
        __bf16* h1out = &sH1[p ^ 1][0];
#pragma unroll
        for (int r = 0; r < 8; ++r) {
            float ig = sigmoid_f(acc[0][r] + b1g[0]);
            float fg = sigmoid_f(acc[1][r] + b1g[1]);
            float gg = tanh_f(acc[2][r] + b1g[2]);
            float og = sigmoid_f(acc[3][r] + b1g[3]);
            float cn = fg * c1[r] + ig * gg;
            c1[r] = cn;
            float hn = og * tanh_f(cn);
            h1out[(r + mrow) * WSTR + wave * 16 + m16] = (__bf16)hn;
        }
        __syncthreads();   // new h1 visible (+ guards staging reuse)

        xf = xfn;
        p ^= 1;
    }

    // ================= head: fc1 -> relu -> fc2 -> sigmoid =================
    // final h1 lives in sH1[p] (16 x 64, padded rows). Lanes 0..15 of wave 0.
    if (wave == 0 && lane < 16) {
        const __bf16* hrow = &sH1[p][m16 * WSTR];
        float acc2 = fc2b[0];
#pragma unroll 4
        for (int j = 0; j < 32; ++j) {
            float s = fc1b[j];
#pragma unroll
            for (int k = 0; k < HDIM; ++k)
                s += (float)hrow[k] * fc1w[j * HDIM + k];
            s = fmaxf(s, 0.0f);
            acc2 += s * fc2w[j];
        }
        out[rowBase + m16] = sigmoid_f(acc2);
    }
}

extern "C" void kernel_launch(void* const* d_in, const int* in_sizes, int n_in,
                              void* d_out, int out_size, void* d_ws, size_t ws_size,
                              hipStream_t stream) {
    const float* x    = (const float*)d_in[0];
    const float* Wih0 = (const float*)d_in[1];
    const float* Whh0 = (const float*)d_in[2];
    const float* bih0 = (const float*)d_in[3];
    const float* bhh0 = (const float*)d_in[4];
    const float* Wih1 = (const float*)d_in[5];
    const float* Whh1 = (const float*)d_in[6];
    const float* bih1 = (const float*)d_in[7];
    const float* bhh1 = (const float*)d_in[8];
    const float* fc1w = (const float*)d_in[9];
    const float* fc1b = (const float*)d_in[10];
    const float* fc2w = (const float*)d_in[11];
    const float* fc2b = (const float*)d_in[12];
    float* out = (float*)d_out;

    const int B = in_sizes[0] / (T_STEPS * D_IN);   // 4096
    dim3 grid(B / 16), block(128);
    lstm2_fused_kernel<<<grid, block, 0, stream>>>(
        x, Wih0, Whh0, bih0, bhh0, Wih1, Whh1, bih1, bhh1,
        fc1w, fc1b, fc2w, fc2b, out);
}
